// VanilaSelfAttention_13168369730002
// MI455X (gfx1250) — matrix operand order repaired and checked by reading, not compiled, not run on hardware
//
#include <hip/hip_runtime.h>

// ---------------------------------------------------------------------------
// CDNA5 (gfx1250) bf16 WMMA self-attention
//   x:(2,4096,1024) f32, Wq/Wk/Wv/Wo:(1024,1024) f32 -> out:(2,4096,1024) f32
// ---------------------------------------------------------------------------

typedef __attribute__((ext_vector_type(16))) __bf16          v16bf;
typedef __attribute__((ext_vector_type(8)))  float           v8f;
typedef __attribute__((ext_vector_type(8)))  unsigned short  v8us;
typedef __attribute__((ext_vector_type(4)))  unsigned short  v4us;
typedef __attribute__((ext_vector_type(16))) unsigned short  v16us;
typedef __attribute__((ext_vector_type(4)))  unsigned int    u32x4;
typedef __attribute__((ext_vector_type(8)))  int             i32x8;
typedef __attribute__((ext_vector_type(4)))  int             i32x4;

static __device__ __forceinline__ unsigned short f2bf(float f) {
  unsigned u = __builtin_bit_cast(unsigned, f);
  u += 0x7FFFu + ((u >> 16) & 1u);               // round-to-nearest-even
  return (unsigned short)(u >> 16);
}

static __device__ __forceinline__ v16bf mkfrag(v8us lo, v8us hi) {
  v16us t;
#pragma unroll
  for (int i = 0; i < 8; ++i) { t[i] = lo[i]; t[i + 8] = hi[i]; }
  return __builtin_bit_cast(v16bf, t);
}

static __device__ __forceinline__ v16bf mkfrag4(v4us a, v4us b, v4us c, v4us d) {
  v16us t;
#pragma unroll
  for (int i = 0; i < 4; ++i) {
    t[i] = a[i]; t[4 + i] = b[i]; t[8 + i] = c[i]; t[12 + i] = d[i];
  }
  return __builtin_bit_cast(v16bf, t);
}

static __device__ __forceinline__ v8f wmma_bf16(v16bf a, v16bf b, v8f c) {
  // (neg_a, A, neg_b, B, c_mod, C, reuse_a, reuse_b)
  return __builtin_amdgcn_wmma_f32_16x16x32_bf16(false, a, false, b,
                                                 (short)0, c, false, false);
}

// ---------------------------------------------------------------------------
// fp32 -> bf16 conversion (grid-stride)
// ---------------------------------------------------------------------------
__global__ void f32_to_bf16_kernel(const float* __restrict__ in,
                                   unsigned short* __restrict__ out, int n) {
  int i = blockIdx.x * blockDim.x + threadIdx.x;
  int stride = gridDim.x * blockDim.x;
  for (; i < n; i += stride) out[i] = f2bf(in[i]);
}

// ---------------------------------------------------------------------------
// Tiled bf16 GEMM:  C = A * Bw^T (Bw row-major [N][K]).  128x128 block tile,
// BK=32, 8 waves (2x4), each wave 4x2 grid of 16x16x32 WMMAs, LDS staged,
// global_prefetch of the next K-tile overlapped with compute.
// ---------------------------------------------------------------------------
template <typename OutT, bool TRANS_OUT>
__global__ __launch_bounds__(256) void gemm_bf16_kernel(
    const unsigned short* __restrict__ A, int lda,
    const unsigned short* __restrict__ Bw, int ldb,
    OutT* __restrict__ C, int ldc,
    int M, int N, int K,
    int rowsPerBatch, long long batchStrideOut) {
  constexpr int LDSW = 40;                       // padded ushort stride
  __shared__ unsigned short As[128 * LDSW];      // 10 KB
  __shared__ unsigned short Bs[128 * LDSW];      // 10 KB

  const int m0   = blockIdx.y * 128;
  const int n0   = blockIdx.x * 128;
  const int t    = threadIdx.x;
  const int lane = t & 31, wave = t >> 5;
  const int wm   = wave & 1, wn = wave >> 1;     // 2 (M) x 4 (N) waves
  const int half = lane >> 4, l16 = lane & 15;

  const int ldrow = t >> 1;                      // 0..127
  const int ldcol = (t & 1) << 4;                // 0 or 16

  v8f acc[4][2];
#pragma unroll
  for (int i = 0; i < 4; ++i)
#pragma unroll
    for (int j = 0; j < 2; ++j) acc[i][j] = v8f{};

  for (int k0 = 0; k0 < K; k0 += 32) {
    // ---- stage A/B tiles into LDS (each thread: 32B of A + 32B of B) ----
    const v8us* ga = reinterpret_cast<const v8us*>(
        A + (size_t)(m0 + ldrow) * lda + k0 + ldcol);
    v8us a0 = ga[0], a1 = ga[1];
    const v8us* gb = reinterpret_cast<const v8us*>(
        Bw + (size_t)(n0 + ldrow) * ldb + k0 + ldcol);
    v8us b0 = gb[0], b1 = gb[1];
    *reinterpret_cast<v8us*>(&As[ldrow * LDSW + ldcol])     = a0;
    *reinterpret_cast<v8us*>(&As[ldrow * LDSW + ldcol + 8]) = a1;
    *reinterpret_cast<v8us*>(&Bs[ldrow * LDSW + ldcol])     = b0;
    *reinterpret_cast<v8us*>(&Bs[ldrow * LDSW + ldcol + 8]) = b1;
    if (k0 + 32 < K) {   // pull next K-tile toward L0/L2 while we compute
      __builtin_prefetch(A + (size_t)(m0 + ldrow) * lda + k0 + 32 + ldcol, 0, 0);
      __builtin_prefetch(Bw + (size_t)(n0 + ldrow) * ldb + k0 + 32 + ldcol, 0, 0);
    }
    __syncthreads();

    // ---- fragments from LDS (ISA 7.12.2 layouts) ----
    v16bf af[4], bfr[2];
#pragma unroll
    for (int i = 0; i < 4; ++i) {
      const unsigned short* r = &As[(wm * 64 + i * 16 + l16) * LDSW];
      af[i] = mkfrag(*reinterpret_cast<const v8us*>(r + half * 8),
                     *reinterpret_cast<const v8us*>(r + 16 + half * 8));
    }
#pragma unroll
    for (int j = 0; j < 2; ++j) {
      const unsigned short* r = &Bs[(wn * 32 + j * 16 + l16) * LDSW];
      bfr[j] = mkfrag(*reinterpret_cast<const v8us*>(r + half * 8),
                      *reinterpret_cast<const v8us*>(r + 16 + half * 8));
    }
#pragma unroll
    for (int i = 0; i < 4; ++i)
#pragma unroll
      for (int j = 0; j < 2; ++j)
        acc[i][j] = wmma_bf16(af[i], bfr[j], acc[i][j]);
    __syncthreads();
  }

  // ---- epilogue ----
#pragma unroll
  for (int i = 0; i < 4; ++i)
#pragma unroll
    for (int j = 0; j < 2; ++j)
#pragma unroll
      for (int r = 0; r < 8; ++r) {
        float v  = acc[i][j][r];
        int   m  = m0 + wm * 64 + i * 16 + half * 8 + r;
        int   nn = n0 + wn * 32 + j * 16 + l16;
        if constexpr (TRANS_OUT) {
          int b  = m / rowsPerBatch;
          int ml = m - b * rowsPerBatch;
          ((unsigned short*)C)[b * batchStrideOut + (size_t)nn * ldc + ml] =
              f2bf(v);
        } else if constexpr (sizeof(OutT) == 4) {
          ((float*)C)[(size_t)m * ldc + nn] = v;
        } else {
          ((unsigned short*)C)[(size_t)m * ldc + nn] = f2bf(v);
        }
      }
}

// ---------------------------------------------------------------------------
// Scores + softmax:  P[16 x 4096] = softmax(q_tile @ k^T / 32), bf16 out.
// The 16x1024 q tile is DMA'd ONCE into LDS by the Tensor Data Mover
// (padded rows: +8B per 1KB => 2064B row stride, conflict-free ds reads),
// then 8 waves each own 512 key columns (32 WMMA tiles x 32 K-steps).
// Score rows (16 x 4096 f32 = 256KB) also live in the 320KB WGP LDS.
// ---------------------------------------------------------------------------
#define QS_BYTES 33024   // 16 rows * 2064B padded stride

// piecewise pad mapping: logical byte pos -> padded LDS byte offset
static __device__ __forceinline__ unsigned qs_off(unsigned pos) {
  return pos + ((pos >> 10) << 3);   // +8B per 1024B segment
}

__global__ __launch_bounds__(256) void attn_softmax_kernel(
    const unsigned short* __restrict__ q,     // [4096][1024] (batch offset)
    const unsigned short* __restrict__ kmat,  // [4096][1024]
    unsigned short* __restrict__ P) {         // [4096][4096]
  extern __shared__ char smem_raw[];
  unsigned short* qs = (unsigned short*)smem_raw;          // padded q tile
  float* S   = (float*)(smem_raw + QS_BYTES);              // 16 * 4096 f32
  float* red = S + 16 * 4096;                              // 16 * 16 f32

  const int m0   = blockIdx.x * 16;
  const int t    = threadIdx.x;
  const int lane = t & 31, wave = t >> 5;
  const int half = lane >> 4, l16 = lane & 15;

  // ---- phase 0: stage q tile (16 x 1024 bf16) into LDS once via TDM ----
#if __has_builtin(__builtin_amdgcn_tensor_load_to_lds)
  if (wave == 0) {
    const unsigned long long gq =
        (unsigned long long)(q + (size_t)m0 * 1024);
    u32x4 g0;
    g0[0] = 1u;                                           // count = 1
    g0[1] = 0u;                                           // lds_addr = 0 (qs)
    g0[2] = (unsigned)(gq & 0xFFFFFFFFull);
    g0[3] = (unsigned)((gq >> 32) & 0x01FFFFFFull) | (2u << 30);  // type=2
    i32x8 g1;
    g1[0] = (1 << 16)      // data_size = 2B
          | (1 << 20)      // pad_enable
          | (7 << 22)      // pad_interval: 256 dwords (1KB)
          | (1 << 25);     // pad_amount: 2 dwords (8B)
    g1[1] = (int)((1024u & 0xFFFFu) << 16);               // tensor_dim0 lo16
    g1[2] = (int)((4096u & 0xFFFFu) << 16);               // dim0 hi | dim1 lo
    g1[3] = (int)(1024u << 16);                           // dim1 hi | tile_dim0
    g1[4] = 16;                                           // tile_dim1 (rows)
    g1[5] = 1024;                                         // dim0_stride lo32
    g1[6] = 0;
    g1[7] = 0;
    i32x4 gz4 = {0, 0, 0, 0};
    i32x8 gz8 = {0, 0, 0, 0, 0, 0, 0, 0};
    // 6-arg toolchain variant: (g0, g1, g2, g3, g4, cpol)
    __builtin_amdgcn_tensor_load_to_lds(g0, g1, gz4, gz4, gz8, 0);
#if __has_builtin(__builtin_amdgcn_s_wait_tensorcnt)
    __builtin_amdgcn_s_wait_tensorcnt(0);
#else
    asm volatile("s_wait_tensorcnt 0x0" ::: "memory");
#endif
  }
#else
  // fallback: cooperative staged copy, identical padded layout
  for (int e = t * 8; e < 16 * 1024; e += 256 * 8) {
    unsigned pos = (unsigned)e * 2;
    const unsigned long long* src =
        (const unsigned long long*)(q + (size_t)m0 * 1024 + e);
    unsigned long long* dst =
        (unsigned long long*)((char*)qs + qs_off(pos));
    dst[0] = src[0];
    dst[1] = src[1];
  }
#endif
  __syncthreads();

  // ---- phase 1: S = q k^T / sqrt(D) via WMMA (A from LDS, B from global) ----
  const char* qsb = (const char*)qs;
  for (int kt = 0; kt < 32; ++kt) {
    const int n0 = wave * 512 + kt * 16;
    v8f acc{};
    for (int kk = 0; kk < 1024; kk += 32) {
      unsigned p0 = qs_off((unsigned)(l16 * 2048 + (kk + half * 8) * 2));
      unsigned p1 = qs_off((unsigned)(l16 * 2048 + (kk + 16 + half * 8) * 2));
      v16bf af = mkfrag4(*reinterpret_cast<const v4us*>(qsb + p0),
                         *reinterpret_cast<const v4us*>(qsb + p0 + 8),
                         *reinterpret_cast<const v4us*>(qsb + p1),
                         *reinterpret_cast<const v4us*>(qsb + p1 + 8));
      const unsigned short* kb =
          kmat + (size_t)(n0 + l16) * 1024 + kk + half * 8;
      v16bf bfr = mkfrag(*reinterpret_cast<const v8us*>(kb),
                         *reinterpret_cast<const v8us*>(kb + 16));
      acc = wmma_bf16(af, bfr, acc);
    }
#pragma unroll
    for (int r = 0; r < 8; ++r)
      S[(half * 8 + r) * 4096 + n0 + l16] = acc[r] * 0.03125f;  // 1/sqrt(1024)
  }
  __syncthreads();

  // ---- phase 2: row softmax (16 threads per row, 256 elems each) ----
  const int row = t >> 4, seg = t & 15;
  float* Sr = S + row * 4096 + seg * 256;
  float mx = -3.4e38f;
  for (int j = 0; j < 256; ++j) mx = fmaxf(mx, Sr[j]);
  red[row * 16 + seg] = mx;
  __syncthreads();
  float rmx = -3.4e38f;
  for (int i = 0; i < 16; ++i) rmx = fmaxf(rmx, red[row * 16 + i]);
  __syncthreads();
  float sum = 0.f;
  for (int j = 0; j < 256; ++j) {
    float e = __expf(Sr[j] - rmx);
    Sr[j] = e;
    sum += e;
  }
  red[row * 16 + seg] = sum;
  __syncthreads();
  float rs = 0.f;
  for (int i = 0; i < 16; ++i) rs += red[row * 16 + i];
  float inv = 1.0f / rs;

  unsigned short* Pr = P + (size_t)(m0 + row) * 4096 + seg * 256;
  for (int j = 0; j < 256; ++j) Pr[j] = f2bf(Sr[j] * inv);
}

// ---------------------------------------------------------------------------
// Host-side orchestration
// ---------------------------------------------------------------------------
extern "C" void kernel_launch(void* const* d_in, const int* in_sizes, int n_in,
                              void* d_out, int out_size, void* d_ws,
                              size_t ws_size, hipStream_t stream) {
  constexpr int  D   = 1024;
  constexpr int  Sq  = 4096;
  constexpr int  B   = 2;
  constexpr int  M   = B * Sq;                   // 8192
  constexpr size_t SM_BYTES =
      QS_BYTES + 16 * 4096 * 4 + 16 * 16 * 4;    // 33KB + 256KB + 1KB

  const float* x  = (const float*)d_in[0];
  const float* Wq = (const float*)d_in[1];
  const float* Wk = (const float*)d_in[2];
  const float* Wv = (const float*)d_in[3];
  const float* Wo = (const float*)d_in[4];
  float* out = (float*)d_out;

  char* ws = (char*)d_ws;
  size_t off = 0;
  auto carve = [&](size_t bytes) -> void* {
    void* p = ws + off;
    off = (off + bytes + 255) & ~(size_t)255;
    return p;
  };
  unsigned short* xb  = (unsigned short*)carve((size_t)M * D * 2);
  unsigned short* wqb = (unsigned short*)carve((size_t)D * D * 2);
  unsigned short* wkb = (unsigned short*)carve((size_t)D * D * 2);
  unsigned short* wvb = (unsigned short*)carve((size_t)D * D * 2);
  unsigned short* wob = (unsigned short*)carve((size_t)D * D * 2);
  unsigned short* qb  = (unsigned short*)carve((size_t)M * D * 2);
  unsigned short* kb  = (unsigned short*)carve((size_t)M * D * 2);
  unsigned short* vTb = (unsigned short*)carve((size_t)M * D * 2);  // [B][D][Sq]
  unsigned short* Pb  = (unsigned short*)carve((size_t)Sq * Sq * 2);  // 1 batch
  unsigned short* aob = (unsigned short*)carve((size_t)M * D * 2);

  // 1) fp32 -> bf16
  f32_to_bf16_kernel<<<2048, 256, 0, stream>>>(x, xb, M * D);
  f32_to_bf16_kernel<<<1024, 256, 0, stream>>>(Wq, wqb, D * D);
  f32_to_bf16_kernel<<<1024, 256, 0, stream>>>(Wk, wkb, D * D);
  f32_to_bf16_kernel<<<1024, 256, 0, stream>>>(Wv, wvb, D * D);
  f32_to_bf16_kernel<<<1024, 256, 0, stream>>>(Wo, wob, D * D);

  // 2) projections: q = x Wq^T, k = x Wk^T, v = x Wv^T (stored transposed)
  dim3 gProj(D / 128, M / 128);
  gemm_bf16_kernel<unsigned short, false><<<gProj, 256, 0, stream>>>(
      xb, D, wqb, D, qb, D, M, D, D, 0, 0);
  gemm_bf16_kernel<unsigned short, false><<<gProj, 256, 0, stream>>>(
      xb, D, wkb, D, kb, D, M, D, D, 0, 0);
  gemm_bf16_kernel<unsigned short, true><<<gProj, 256, 0, stream>>>(
      xb, D, wvb, D, vTb, Sq, M, D, D, Sq, (long long)D * Sq);

  // 3) per-batch: P = softmax(q k^T / 32);  ao = P v
  for (int b = 0; b < B; ++b) {
    const unsigned short* qbb  = qb + (size_t)b * Sq * D;
    const unsigned short* kbb  = kb + (size_t)b * Sq * D;
    const unsigned short* vTbb = vTb + (size_t)b * D * Sq;
    unsigned short* aobb = aob + (size_t)b * Sq * D;
    attn_softmax_kernel<<<Sq / 16, 256, SM_BYTES, stream>>>(qbb, kbb, Pb);
    dim3 gPV(D / 128, Sq / 128);
    gemm_bf16_kernel<unsigned short, false><<<gPV, 256, 0, stream>>>(
        Pb, Sq, vTbb, Sq, aobb, D, Sq, D, Sq, 0, 0);
  }

  // 4) output projection, f32 store directly to d_out
  gemm_bf16_kernel<float, false><<<gProj, 256, 0, stream>>>(
      aob, D, wob, D, out, D, M, D, D, 0, 0);
}
